// SlotAttention_60653528154636
// MI455X (gfx1250) — matrix operand order repaired and checked by reading, not compiled, not run on hardware
//
#include <hip/hip_runtime.h>

typedef __attribute__((ext_vector_type(16))) _Float16 v16h;
typedef __attribute__((ext_vector_type(8)))  float    v8f;

#define NB   8
#define NS   8
#define HWN  4096
#define PIX  128
#define SA   72     // padded LDS stride (halves) for t tile  (64 + 8)
#define SH   136    // padded LDS stride (halves) for h tile  (128 + 8)
#define GSTEP (2.0f/63.0f)

// ---------------------------------------------------------------------------
// A-fragment loader: contiguous K order (matches packed B fragments)
// lanes 0-15: K = 0..15   (M = lane)
// lanes 16-31: K = 16..31 (M = lane-16)
// ---------------------------------------------------------------------------
__device__ __forceinline__ v16h load_a_frag(const _Float16* lds, int stride,
                                            int mtile, int kb, int lane) {
  int m = mtile * 16 + (lane & 15);
  int kbase = kb * 32 + ((lane & 16) ? 16 : 0);
  const _Float16* p = lds + m * stride + kbase;
  v16h a;
#pragma unroll
  for (int v = 0; v < 4; ++v) { a[2*v]   = p[2*v];      a[2*v+1]   = p[2*v+1]; }
#pragma unroll
  for (int v = 0; v < 4; ++v) { a[8+2*v] = p[8+2*v];   a[8+2*v+1] = p[8+2*v+1]; }
  return a;
}

__device__ __forceinline__ v8f wmma16(v16h a, v16h b, v8f c) {
  return __builtin_amdgcn_wmma_f32_16x16x32_f16(false, a, false, b, short(0), c,
                                                false, false);
}

// ---------------------------------------------------------------------------
// Fused proj_kv pass.  MODE 0: dots = (q . (MLP(LN(base+pe)) + b2)) * scale
//                      MODE 1: updates += attn . (MLP(...)) (b2 added in GRU)
// One block = 128 pixels of one (b,s).  8 waves.
// ---------------------------------------------------------------------------
template <int MODE>
__global__ __launch_bounds__(256) void proj_pass(
    const float* __restrict__ base,      // k0 or v0  (B*HW, 64)
    const float* __restrict__ Wpos, const float* __restrict__ bpos,
    const float* __restrict__ gain, const float* __restrict__ beta,
    const _Float16* __restrict__ W1F, const float* __restrict__ b1,
    const _Float16* __restrict__ W2F,
    const float* __restrict__ sp, const float* __restrict__ ss,
    const float* __restrict__ qbuf, const float* __restrict__ qb2,   // MODE 0
    const float* __restrict__ mask, const float* __restrict__ invden, // MODE 1
    float* __restrict__ dots,            // MODE 0 output
    float* __restrict__ updates)         // MODE 1 output (atomic)
{
  __shared__ _Float16 tA[PIX * SA];
  __shared__ _Float16 hB[PIX * SH];
  __shared__ float dotsP[8 * PIX];
  __shared__ float updP[8 * 16];
  __shared__ float attnq[PIX];
  __shared__ float wc[5 * 64];   // Wpos row0, Wpos row1, bpos, gain, beta

  const int blk  = blockIdx.x;
  const int tile = blk & 31;          // HW / PIX = 32 tiles
  const int bs   = blk >> 5;          // 0..63
  const int p0   = tile * PIX;
  const int tid  = threadIdx.x;
  const int lane = tid & 31;
  const int w    = tid >> 5;

  for (int i = tid; i < 320; i += 256) {
    int seg = i >> 6, d = i & 63;
    float v = (seg == 0) ? Wpos[d] : (seg == 1) ? Wpos[64 + d]
            : (seg == 2) ? bpos[d] : (seg == 3) ? gain[d] : beta[d];
    wc[i] = v;
  }
  for (int i = tid; i < 8 * PIX; i += 256) dotsP[i] = 0.0f;
  if (MODE == 0) {
    if (tid < 64) attnq[tid] = qbuf[bs * 64 + tid];
  } else {
    if (tid < PIX) attnq[tid] = mask[(size_t)bs * HWN + p0 + tid] * invden[bs];
  }
  __syncthreads();

  // ---- build t = LN(base + pe) in f16, one row per thread (t < 128) ----
  if (tid < PIX) {
    int p = p0 + tid;
    float gx = -1.0f + (float)(p & 63) * GSTEP;
    float gy = -1.0f + (float)(p >> 6) * GSTEP;
    float spx = sp[bs * 2 + 0], spy = sp[bs * 2 + 1];
    float i0 = 1.0f / (ss[bs * 2 + 0] * 5.0f);
    float i1 = 1.0f / (ss[bs * 2 + 1] * 5.0f);
    float g0 = (gx - spx) * i0, g1 = (gy - spy) * i1;
    const float* bp = base + ((size_t)(bs >> 3) * HWN + p) * 64;
    float tv[64];
    float mean = 0.0f;
#pragma unroll
    for (int d = 0; d < 64; ++d) {
      float v = bp[d] + g0 * wc[d] + g1 * wc[64 + d] + wc[128 + d];
      tv[d] = v; mean += v;
    }
    mean *= (1.0f / 64.0f);
    float var = 0.0f;
#pragma unroll
    for (int d = 0; d < 64; ++d) { float c = tv[d] - mean; var += c * c; }
    var *= (1.0f / 64.0f);
    float rs = rsqrtf(var + 1e-5f);
#pragma unroll
    for (int d = 0; d < 64; ++d)
      tA[tid * SA + d] = (_Float16)((tv[d] - mean) * rs * wc[192 + d] + wc[256 + d]);
  }
  __syncthreads();

  // ---- GEMM1: h = relu(t @ W1 + b1), wave w owns N-tile w ----
  {
    const int nt = w;
    v16h bf0 = *(const v16h*)(W1F + ((nt * 2 + 0) * 32 + lane) * 16);
    v16h bf1 = *(const v16h*)(W1F + ((nt * 2 + 1) * 32 + lane) * 16);
    const int n0 = nt * 16 + (lane & 15);
    const float bias = b1[n0];
#pragma unroll
    for (int mt = 0; mt < 8; ++mt) {
      v8f c = {};
      c = wmma16(load_a_frag(tA, SA, mt, 0, lane), bf0, c);
      c = wmma16(load_a_frag(tA, SA, mt, 1, lane), bf1, c);
      const int mB = mt * 16 + ((lane & 16) ? 8 : 0);
#pragma unroll
      for (int r = 0; r < 8; ++r) {
        float hv = c[r] + bias;
        hB[(mB + r) * SH + n0] = (_Float16)fmaxf(hv, 0.0f);
      }
    }
  }
  __syncthreads();

  // ---- GEMM2: out = h @ W2, fused contraction epilogue ----
  {
    const int nt = w & 3;
    const int mg = w >> 2;
    v16h bf[4];
#pragma unroll
    for (int kb = 0; kb < 4; ++kb)
      bf[kb] = *(const v16h*)(W2F + ((nt * 4 + kb) * 32 + lane) * 16);
    float laneAcc = 0.0f;
#pragma unroll
    for (int mi = 0; mi < 4; ++mi) {
      const int mt = mg * 4 + mi;
      v8f c = {};
#pragma unroll
      for (int kb = 0; kb < 4; ++kb)
        c = wmma16(load_a_frag(hB, SH, mt, kb, lane), bf[kb], c);
      const int mB = mt * 16 + ((lane & 16) ? 8 : 0);
      if (MODE == 0) {
        float qv = attnq[nt * 16 + (lane & 15)];
#pragma unroll
        for (int r = 0; r < 8; ++r) {
          float acc = qv * c[r];
          acc += __shfl_xor(acc, 1);
          acc += __shfl_xor(acc, 2);
          acc += __shfl_xor(acc, 4);
          acc += __shfl_xor(acc, 8);
          if ((lane & 15) == 0) dotsP[w * PIX + mB + r] = acc;
        }
      } else {
#pragma unroll
        for (int r = 0; r < 8; ++r) laneAcc += attnq[mB + r] * c[r];
      }
    }
    if (MODE == 1) {
      laneAcc += __shfl_xor(laneAcc, 16);
      if (lane < 16) updP[w * 16 + lane] = laneAcc;
    }
  }
  __syncthreads();

  if (MODE == 0) {
    if (tid < PIX) {
      float v = qb2[bs];
#pragma unroll
      for (int ww = 0; ww < 8; ++ww) v += dotsP[ww * PIX + tid];
      dots[(size_t)bs * HWN + p0 + tid] = v * 0.125f;   // D^-0.5 = 1/8
    }
  } else {
    if (tid < 64) {
      int nt = tid >> 4, i = tid & 15;
      float v = updP[nt * 16 + i] + updP[(nt + 4) * 16 + i];
      atomicAdd(&updates[bs * 64 + tid], v);
    }
  }
}

// ---------------------------------------------------------------------------
// Precompute: x = LN(inputs); k0 = x@Wk+bk; v0 = x@Wv+bv     (one pixel/thread)
// ---------------------------------------------------------------------------
__global__ __launch_bounds__(256) void pre_kernel(
    const float* __restrict__ inputs, const float* __restrict__ g_in,
    const float* __restrict__ b_in, const float* __restrict__ Wk,
    const float* __restrict__ bk, const float* __restrict__ Wv,
    const float* __restrict__ bv, float* __restrict__ k0, float* __restrict__ v0)
{
  int p = blockIdx.x * 256 + threadIdx.x;          // 0 .. B*HW-1
  const float* row = inputs + (size_t)p * 64;
  float x[64];
  float mean = 0.0f;
#pragma unroll
  for (int d = 0; d < 64; ++d) { x[d] = row[d]; mean += x[d]; }
  mean *= (1.0f / 64.0f);
  float var = 0.0f;
#pragma unroll
  for (int d = 0; d < 64; ++d) { float c = x[d] - mean; var += c * c; }
  var *= (1.0f / 64.0f);
  float rs = rsqrtf(var + 1e-5f);
#pragma unroll
  for (int d = 0; d < 64; ++d) x[d] = (x[d] - mean) * rs * g_in[d] + b_in[d];
  for (int n = 0; n < 64; ++n) {
    float ak = bk[n], av = bv[n];
#pragma unroll
    for (int d = 0; d < 64; ++d) { ak += x[d] * Wk[d * 64 + n]; av += x[d] * Wv[d * 64 + n]; }
    k0[(size_t)p * 64 + n] = ak;
    v0[(size_t)p * 64 + n] = av;
  }
}

// ---------------------------------------------------------------------------
// Pack W1/W2 into per-lane WMMA B-fragment layout (contiguous 16 halves/lane).
// B 32x16 k order used here (contiguous, same as A): lanes 0-15 K=0..15, lanes 16-31 K=16..31, N = lane&15.
// ---------------------------------------------------------------------------
__global__ void pack_kernel(const float* __restrict__ W1,
                            const float* __restrict__ W2,
                            _Float16* __restrict__ W1F, _Float16* __restrict__ W2F)
{
  int u = blockIdx.x * 256 + threadIdx.x;
  if (u < 512) {                     // W1 (64x128): 8 n-tiles x 2 k-blocks
    int f = u >> 5, L = u & 31, nt = f >> 1, kb = f & 1;
    int k0 = kb * 32 + ((L & 16) ? 16 : 0);
    int n  = nt * 16 + (L & 15);
#pragma unroll
    for (int t = 0; t < 16; ++t) W1F[u * 16 + t] = (_Float16)W1[(k0 + t) * 128 + n];
  } else if (u < 1024) {             // W2 (128x64): 4 n-tiles x 4 k-blocks
    int v = u - 512;
    int f = v >> 5, L = v & 31, nt = f >> 2, kb = f & 3;
    int k0 = kb * 32 + ((L & 16) ? 16 : 0);
    int n  = nt * 16 + (L & 15);
#pragma unroll
    for (int t = 0; t < 16; ++t) W2F[v * 16 + t] = (_Float16)W2[(k0 + t) * 64 + n];
  }
}

// ---------------------------------------------------------------------------
__global__ void init_kernel(const float* __restrict__ slots_p,
                            const float* __restrict__ s_pos_p,
                            const float* __restrict__ s_scale_p,
                            float* __restrict__ slots, float* __restrict__ sp,
                            float* __restrict__ ss)
{
  int t = blockIdx.x * 256 + threadIdx.x;
  if (t < 4096) slots[t] = slots_p[t & 511];
  if (t < 128) {
    sp[t] = fminf(fmaxf(s_pos_p[t & 15], -1.0f), 1.0f);
    ss[t] = fminf(fmaxf(s_scale_p[t & 15], 0.001f), 2.0f);
  }
}

// q = LN(slots)@Wq + bq ; qb2[row] = q . b2
__global__ __launch_bounds__(64) void q_kernel(
    const float* __restrict__ slots, const float* __restrict__ g_s,
    const float* __restrict__ b_s, const float* __restrict__ Wq,
    const float* __restrict__ bq, const float* __restrict__ b2,
    float* __restrict__ q, float* __restrict__ qb2)
{
  int row = threadIdx.x;
  const float* sr = slots + row * 64;
  float x[64];
  float mean = 0.0f;
#pragma unroll
  for (int d = 0; d < 64; ++d) { x[d] = sr[d]; mean += x[d]; }
  mean *= (1.0f / 64.0f);
  float var = 0.0f;
#pragma unroll
  for (int d = 0; d < 64; ++d) { float c = x[d] - mean; var += c * c; }
  var *= (1.0f / 64.0f);
  float rs = rsqrtf(var + 1e-5f);
#pragma unroll
  for (int d = 0; d < 64; ++d) x[d] = (x[d] - mean) * rs * g_s[d] + b_s[d];
  float acc2 = 0.0f;
  for (int n = 0; n < 64; ++n) {
    float a = bq[n];
#pragma unroll
    for (int d = 0; d < 64; ++d) a += x[d] * Wq[d * 64 + n];
    q[row * 64 + n] = a;
    acc2 += a * b2[n];
  }
  qb2[row] = acc2;
}

// softmax over slots per pixel; + EPS; in-place dots -> mask
__global__ void softmax_kernel(float* __restrict__ dots,
                               float* __restrict__ out_attn, int last)
{
  int g = blockIdx.x * 256 + threadIdx.x;   // B*HW entries
  if (g >= NB * HWN) return;
  int b = g >> 12, j = g & 4095;
  float d[NS];
  float mx = -1e30f;
#pragma unroll
  for (int s = 0; s < NS; ++s) {
    d[s] = dots[((size_t)(b * NS + s)) * HWN + j];
    mx = fmaxf(mx, d[s]);
  }
  float sum = 0.0f;
#pragma unroll
  for (int s = 0; s < NS; ++s) { d[s] = expf(d[s] - mx); sum += d[s]; }
  float inv = 1.0f / sum;
#pragma unroll
  for (int s = 0; s < NS; ++s) {
    float m = d[s] * inv + 1e-8f;
    dots[((size_t)(b * NS + s)) * HWN + j] = m;
    if (last) out_attn[((size_t)(b * NS + s)) * HWN + j] = m;
  }
}

// per-(b,s): sp, ss, inv_denom; zero updates; optional final outputs
__global__ __launch_bounds__(256) void reduce_kernel(
    const float* __restrict__ mask, float* __restrict__ sp_nxt,
    float* __restrict__ ss_nxt, float* __restrict__ invden,
    float* __restrict__ updates, float* __restrict__ out_sp,
    float* __restrict__ out_ss, int last)
{
  __shared__ float r0[256], r1[256], r2[256];
  int bs = blockIdx.x, t = threadIdx.x;
  float sm = 0.0f, sx = 0.0f, sy = 0.0f;
  for (int j = t; j < HWN; j += 256) {
    float m = mask[(size_t)bs * HWN + j];
    float gx = -1.0f + (float)(j & 63) * GSTEP;
    float gy = -1.0f + (float)(j >> 6) * GSTEP;
    sm += m; sx += m * gx; sy += m * gy;
  }
  r0[t] = sm; r1[t] = sx; r2[t] = sy;
  __syncthreads();
  for (int o = 128; o > 0; o >>= 1) {
    if (t < o) { r0[t] += r0[t + o]; r1[t] += r1[t + o]; r2[t] += r2[t + o]; }
    __syncthreads();
  }
  float smT = r0[0];
  float spx = r1[0] / smT, spy = r2[0] / smT;
  __syncthreads();

  float sa = 0.0f, dx = 0.0f, dy = 0.0f;
  for (int j = t; j < HWN; j += 256) {
    float m = mask[(size_t)bs * HWN + j] + 1e-8f;
    float gx = -1.0f + (float)(j & 63) * GSTEP;
    float gy = -1.0f + (float)(j >> 6) * GSTEP;
    float cx = gx - spx, cy = gy - spy;
    sa += m; dx += m * cx * cx; dy += m * cy * cy;
  }
  r0[t] = sa; r1[t] = dx; r2[t] = dy;
  __syncthreads();
  for (int o = 128; o > 0; o >>= 1) {
    if (t < o) { r0[t] += r0[t + o]; r1[t] += r1[t + o]; r2[t] += r2[t + o]; }
    __syncthreads();
  }
  if (t == 0) {
    float invA = 1.0f / r0[0];
    float sxx = fminf(fmaxf(sqrtf(r1[0] * invA), 0.001f), 2.0f);
    float syy = fminf(fmaxf(sqrtf(r2[0] * invA), 0.001f), 2.0f);
    sp_nxt[bs * 2 + 0] = spx; sp_nxt[bs * 2 + 1] = spy;
    ss_nxt[bs * 2 + 0] = sxx; ss_nxt[bs * 2 + 1] = syy;
    invden[bs] = 1.0f / smT;
    if (last) {
      out_sp[bs * 2 + 0] = spx; out_sp[bs * 2 + 1] = spy;
      out_ss[bs * 2 + 0] = sxx; out_ss[bs * 2 + 1] = syy;
    }
  }
  if (t < 64) updates[bs * 64 + t] = 0.0f;   // zeroed before updates pass
}

// GRU + pre-LN MLP on 64 slot rows (single block)
__global__ __launch_bounds__(256) void gru_kernel(
    float* __restrict__ slots, const float* __restrict__ updates,
    const float* __restrict__ b2, const float* __restrict__ Wih,
    const float* __restrict__ bih, const float* __restrict__ Whh,
    const float* __restrict__ bhh, const float* __restrict__ g_pre,
    const float* __restrict__ b_pre, const float* __restrict__ Wf1,
    const float* __restrict__ bf1, const float* __restrict__ Wf2,
    const float* __restrict__ bf2)
{
  __shared__ float buf[16384];
  float* sNew = buf;
  float* sPrev = buf + 4096;
  int t = threadIdx.x;
  for (int i = t; i < 4096; i += 256) sPrev[i] = slots[i];
  __syncthreads();
  for (int k = 0; k < 16; ++k) {
    int idx = t + k * 256, row = idx >> 6, d = idx & 63;
    float ir = bih[d], iz = bih[64 + d], in_ = bih[128 + d];
    float hr = bhh[d], hz = bhh[64 + d], hn = bhh[128 + d];
    const float* up = updates + row * 64;
    const float* pv = sPrev + row * 64;
    for (int c = 0; c < 64; ++c) {
      float u = up[c] + b2[c];
      float p = pv[c];
      ir += u * Wih[c * 192 + d];       hr += p * Whh[c * 192 + d];
      iz += u * Wih[c * 192 + 64 + d];  hz += p * Whh[c * 192 + 64 + d];
      in_ += u * Wih[c * 192 + 128 + d]; hn += p * Whh[c * 192 + 128 + d];
    }
    float r = 1.0f / (1.0f + expf(-(ir + hr)));
    float z = 1.0f / (1.0f + expf(-(iz + hz)));
    float n = tanhf(in_ + r * hn);
    sNew[idx] = (1.0f - z) * n + z * sPrev[idx];
  }
  __syncthreads();
  float* sLn = buf + 4096;   // overwrites sPrev (no longer needed)
  if (t < 64) {
    float mean = 0.0f;
    for (int d = 0; d < 64; ++d) mean += sNew[t * 64 + d];
    mean *= (1.0f / 64.0f);
    float var = 0.0f;
    for (int d = 0; d < 64; ++d) { float c = sNew[t * 64 + d] - mean; var += c * c; }
    var *= (1.0f / 64.0f);
    float rs = rsqrtf(var + 1e-5f);
    for (int d = 0; d < 64; ++d)
      sLn[t * 64 + d] = (sNew[t * 64 + d] - mean) * rs * g_pre[d] + b_pre[d];
  }
  __syncthreads();
  float* ffh = buf + 8192;
  for (int k = 0; k < 32; ++k) {
    int idx = t + k * 256, row = idx >> 7, c = idx & 127;
    float a = bf1[c];
    for (int d = 0; d < 64; ++d) a += sLn[row * 64 + d] * Wf1[d * 128 + c];
    ffh[idx] = fmaxf(a, 0.0f);
  }
  __syncthreads();
  for (int k = 0; k < 16; ++k) {
    int idx = t + k * 256, row = idx >> 6, d = idx & 63;
    float a = bf2[d];
    for (int c = 0; c < 128; ++c) a += ffh[row * 128 + c] * Wf2[c * 64 + d];
    slots[idx] = sNew[idx] + a;
  }
}

__global__ void copyslots_kernel(const float* __restrict__ slots,
                                 float* __restrict__ out)
{
  int t = blockIdx.x * 256 + threadIdx.x;
  if (t < 4096) out[t] = slots[t];
}

// ---------------------------------------------------------------------------
extern "C" void kernel_launch(void* const* d_in, const int* in_sizes, int n_in,
                              void* d_out, int out_size, void* d_ws, size_t ws_size,
                              hipStream_t stream) {
  // setup_inputs() dict order
  const float* inputs   = (const float*)d_in[0];
  const float* slots_p  = (const float*)d_in[1];
  const float* s_pos_p  = (const float*)d_in[2];
  const float* s_scale_p= (const float*)d_in[3];
  const float* W_pos    = (const float*)d_in[4];
  const float* b_pos    = (const float*)d_in[5];
  const float* Wq       = (const float*)d_in[6];
  const float* bq       = (const float*)d_in[7];
  const float* Wk       = (const float*)d_in[8];
  const float* bk       = (const float*)d_in[9];
  const float* Wv       = (const float*)d_in[10];
  const float* bv       = (const float*)d_in[11];
  const float* g_in     = (const float*)d_in[12];
  const float* b_in     = (const float*)d_in[13];
  const float* g_slots  = (const float*)d_in[14];
  const float* b_slots  = (const float*)d_in[15];
  const float* g_k      = (const float*)d_in[16];
  const float* b_k      = (const float*)d_in[17];
  const float* g_v      = (const float*)d_in[18];
  const float* b_v      = (const float*)d_in[19];
  const float* W1       = (const float*)d_in[20];
  const float* b1       = (const float*)d_in[21];
  const float* W2       = (const float*)d_in[22];
  const float* b2       = (const float*)d_in[23];
  const float* W_ih     = (const float*)d_in[24];
  const float* b_ih     = (const float*)d_in[25];
  const float* W_hh     = (const float*)d_in[26];
  const float* b_hh     = (const float*)d_in[27];
  const float* g_pre    = (const float*)d_in[28];
  const float* b_pre    = (const float*)d_in[29];
  const float* Wf1      = (const float*)d_in[30];
  const float* bf1      = (const float*)d_in[31];
  const float* Wf2      = (const float*)d_in[32];
  const float* bf2      = (const float*)d_in[33];

  float* ws = (float*)d_ws;
  float* k0    = ws;                   // 2,097,152
  float* v0    = k0 + 2097152;         // 2,097,152
  float* dots  = v0 + 2097152;         // 262,144 (reused as attn_mask)
  float* qbuf  = dots + 262144;        // 4096
  float* qb2   = qbuf + 4096;          // 64
  float* slots = qb2 + 64;             // 4096
  float* upd   = slots + 4096;         // 4096
  float* spA   = upd + 4096;           // 128
  float* ssA   = spA + 128;            // 128
  float* spB   = ssA + 128;            // 128
  float* ssB   = spB + 128;            // 128
  float* invd  = ssB + 128;            // 64
  _Float16* W1F = (_Float16*)(invd + 64);          // 8192 halves
  _Float16* W2F = (_Float16*)(invd + 64 + 4096);   // 8192 halves

  float* outf     = (float*)d_out;
  float* out_sp   = outf + 4096;
  float* out_ss   = outf + 4096 + 128;
  float* out_attn = outf + 4096 + 256;

  pre_kernel<<<128, 256, 0, stream>>>(inputs, g_in, b_in, Wk, bk, Wv, bv, k0, v0);
  pack_kernel<<<4, 256, 0, stream>>>(W1, W2, W1F, W2F);
  init_kernel<<<16, 256, 0, stream>>>(slots_p, s_pos_p, s_scale_p, slots, spA, ssA);

  float *sp_cur = spA, *ss_cur = ssA, *sp_nxt = spB, *ss_nxt = ssB;
  for (int it = 0; it <= 3; ++it) {
    int last = (it == 3);
    q_kernel<<<1, 64, 0, stream>>>(slots, g_slots, b_slots, Wq, bq, b2, qbuf, qb2);

    proj_pass<0><<<2048, 256, 0, stream>>>(k0, W_pos, b_pos, g_k, b_k, W1F, b1,
                                           W2F, sp_cur, ss_cur, qbuf, qb2,
                                           nullptr, nullptr, dots, nullptr);
    softmax_kernel<<<128, 256, 0, stream>>>(dots, out_attn, last);
    reduce_kernel<<<64, 256, 0, stream>>>(dots, sp_nxt, ss_nxt, invd, upd,
                                          out_sp, out_ss, last);
    if (!last) {
      proj_pass<1><<<2048, 256, 0, stream>>>(v0, W_pos, b_pos, g_v, b_v, W1F, b1,
                                             W2F, sp_cur, ss_cur, nullptr, nullptr,
                                             dots, invd, nullptr, upd);
      gru_kernel<<<1, 256, 0, stream>>>(slots, upd, b2, W_ih, b_ih, W_hh, b_hh,
                                        g_pre, b_pre, Wf1, bf1, Wf2, bf2);
    }
    float* tp;
    tp = sp_cur; sp_cur = sp_nxt; sp_nxt = tp;
    tp = ss_cur; ss_cur = ss_nxt; ss_nxt = tp;
  }
  copyslots_kernel<<<16, 256, 0, stream>>>(slots, outf);
}
